// GroupedMultiQueryAttention_12034498363675
// MI455X (gfx1250) — matrix operand order repaired and checked by reading, not compiled, not run on hardware
//
#include <hip/hip_runtime.h>
#include <cstdint>

// ---------------------------------------------------------------------------
// Types matching CDNA5 WMMA / TDM builtin signatures
// ---------------------------------------------------------------------------
typedef __bf16 bf16_t;
typedef __attribute__((ext_vector_type(16))) __bf16 v16bf;
typedef __attribute__((ext_vector_type(8)))  __bf16 v8bf;
typedef __attribute__((ext_vector_type(8)))  float  v8f;
typedef __attribute__((ext_vector_type(4))) unsigned int u32x4;
typedef __attribute__((ext_vector_type(8))) int         i32x8;
typedef __attribute__((ext_vector_type(4))) int         i32x4;

union V16U { v16bf v; v8bf h[2]; };

#define WMMA_BF16(A, B, C) \
    __builtin_amdgcn_wmma_f32_16x16x32_bf16(false, (A), false, (B), (short)0, (C), false, false)

// Constants of the problem
#define T_SEQ   2048
#define HID     2048
#define NH      32
#define NKV     8
#define HD      64
#define KVDIM   (NKV * HD)   // 512

// ---------------------------------------------------------------------------
// TDM availability (this toolchain: 6-arg builtin, confirmed last round)
// ---------------------------------------------------------------------------
#if __has_builtin(__builtin_amdgcn_tensor_load_to_lds)
#define HAVE_TDM 1
#else
#define HAVE_TDM 0
#endif

#if __has_builtin(__builtin_amdgcn_s_wait_tensorcnt)
#define WAIT_TENSOR(N) __builtin_amdgcn_s_wait_tensorcnt(N)
#else
#define WAIT_TENSOR(N) ((void)0)
#endif

#if HAVE_TDM
// Issue a TDM 2-D tile load: tile_d1 rows x tile_d0 elements (bf16) from a
// row-major tensor (row stride = stride0 elements) into LDS at lds_off,
// packed linearly [tile_d1][tile_d0]. D# bit layout per CDNA5 ISA ch.8.
static __device__ inline void tdm_load_2d(const bf16_t* gsrc, unsigned lds_off,
                                          int tensor_d0, int tensor_d1, int stride0,
                                          int tile_d0, int tile_d1) {
    unsigned long long ga = (unsigned long long)(uintptr_t)gsrc;
    u32x4 g0;
    g0[0] = 1u;                                            // count=1, user mode
    g0[1] = lds_off;                                       // lds_addr (bytes)
    g0[2] = (unsigned)(ga & 0xFFFFFFFFu);                  // global_addr[31:0]
    g0[3] = (unsigned)((ga >> 32) & 0x01FFFFFFu) | (2u << 30);  // addr[56:32] | type=2
    i32x8 g1;
    g1[0] = (1 << 16);                                     // data_size=1 -> 2 bytes
    g1[1] = (tensor_d0 & 0xFFFF) << 16;                    // tensor_dim0[15:0] @bit48
    g1[2] = ((tensor_d0 >> 16) & 0xFFFF) | ((tensor_d1 & 0xFFFF) << 16);
    g1[3] = ((tensor_d1 >> 16) & 0xFFFF) | ((tile_d0 & 0xFFFF) << 16);
    g1[4] = (tile_d1 & 0xFFFF);                            // tile_dim1, tile_dim2=0
    g1[5] = stride0;                                       // tensor_dim0_stride[31:0]
    g1[6] = 0;
    g1[7] = 0;
    i32x4 z4 = {0, 0, 0, 0};                               // groups 2/3 unused (2-D)
    i32x8 z8 = {0, 0, 0, 0, 0, 0, 0, 0};
    __builtin_amdgcn_tensor_load_to_lds(g0, g1, z4, z4, z8, 0);
}
#endif

// ---------------------------------------------------------------------------
// Register tile loaders (16-bit WMMA operand layouts, wave32)
// ---------------------------------------------------------------------------
static __device__ inline v16bf ld_a_tile(const bf16_t* __restrict__ src, int ld,
                                         int r0, int k0, int lane) {
    const int m  = lane & 15;
    const int hf = lane >> 4;
    const bf16_t* p = src + (size_t)(r0 + m) * ld + k0;
    V16U u;
    u.h[0] = *reinterpret_cast<const v8bf*>(p + hf * 16);
    u.h[1] = *reinterpret_cast<const v8bf*>(p + hf * 16 + 8);
    return u.v;
}

static __device__ inline v16bf ld_b_tile(const bf16_t* __restrict__ src, int ld,
                                         int n0, int k0, int lane) {
    const int n  = lane & 15;
    const int hf = lane >> 4;
    const bf16_t* p = src + (size_t)(n0 + n) * ld + k0 + hf * 16;
    return *reinterpret_cast<const v16bf*>(p);
}

// B tile read out of LDS staging buffer laid out [64 rows][32 cols] bf16
static __device__ inline v16bf ld_b_lds(const bf16_t* base, int n0, int lane) {
    const int n  = lane & 15;
    const int hf = lane >> 4;
    const bf16_t* p = base + (n0 + n) * 32 + hf * 16;
    V16U u;
    u.h[0] = *reinterpret_cast<const v8bf*>(p);
    u.h[1] = *reinterpret_cast<const v8bf*>(p + 8);
    return u.v;
}

// ---------------------------------------------------------------------------
// fp32 -> bf16 cast
// ---------------------------------------------------------------------------
__global__ void cast_f32_bf16(const float* __restrict__ in, bf16_t* __restrict__ out, int nel) {
    int i = blockIdx.x * blockDim.x + threadIdx.x;
    if (i < nel) out[i] = (bf16_t)in[i];
}

// ---------------------------------------------------------------------------
// WMMA GEMM: y[M,N] = A[M,K] * W[N,K]^T  (bf16 in, f32 accum)
// 128-thread block (4 waves): block tile 64M x 64N; each wave owns 16M x 64N.
// B tile (64x32 bf16, 4KB) shared by all 4 waves, staged in LDS double-buffer
// by the Tensor Data Mover (TENSORcnt), with a single uniform
// s_wait_tensorcnt 1 per iteration (next tile always in flight; last
// iteration re-issues the final tile as a harmless dummy).
// A operand is software-pipelined in registers; all 4 B tiles are read from
// LDS before the WMMA chain so the 4 matrix ops issue back-to-back.
// mode: 0 = bf16 out, 1 = bf16 transposed out, 2 = f32 out.
// ---------------------------------------------------------------------------
__global__ __launch_bounds__(128)
void wmma_gemm_bf16(const bf16_t* __restrict__ A, const bf16_t* __restrict__ W,
                    void* __restrict__ outp, int K, int ldA, int ldW, int ldO,
                    int NW, int mode) {
    const int tid  = threadIdx.x;
    const int lane = tid & 31;
    const int wv   = tid >> 5;                 // wave id 0..3
    const int r0   = blockIdx.x * 64 + wv * 16;
    const int c0   = blockIdx.y * 64;

    __shared__ alignas(16) bf16_t Bsm[2][64 * 32];

    const int nk = K / 32;
    v8f acc[4] = {};

    // ---- stage first B tile into buffer 0
#if HAVE_TDM
    if (wv == 0) {
        tdm_load_2d(W + (size_t)c0 * ldW, (unsigned)(uintptr_t)&Bsm[0][0],
                    K, NW, ldW, 32, 64);
    }
#else
    for (int t = tid; t < 256; t += 128) {     // 64 rows x 4 chunks of 8 bf16
        int row = t >> 2, ch = t & 3;
        *reinterpret_cast<v8bf*>(&Bsm[0][row * 32 + ch * 8]) =
            *reinterpret_cast<const v8bf*>(&W[(size_t)(c0 + row) * ldW + ch * 8]);
    }
#endif

    v16bf a_cur = ld_a_tile(A, ldA, r0, 0, lane);

    for (int it = 0; it < nk; ++it) {
        const int k0 = it * 32;
        const bool more = (it + 1 < nk);
        // clamp so the last iteration issues a harmless re-load of the final
        // tile -> in-flight invariant is always 2 and the wait is uniform
        const int knext = (more ? (it + 1) : it) * 32;

#if HAVE_TDM
        if (wv == 0) {
            tdm_load_2d(W + (size_t)c0 * ldW + knext,
                        (unsigned)(uintptr_t)&Bsm[(it + 1) & 1][0],
                        K, NW, ldW, 32, 64);
            WAIT_TENSOR(1);                    // current buffer complete
        }
#else
        for (int t = tid; t < 256; t += 128) {
            int row = t >> 2, ch = t & 3;
            *reinterpret_cast<v8bf*>(&Bsm[(it + 1) & 1][row * 32 + ch * 8]) =
                *reinterpret_cast<const v8bf*>(
                    &W[(size_t)(c0 + row) * ldW + knext + ch * 8]);
        }
#endif
        __syncthreads();

        // software-pipeline the A operand for the next k-step
        v16bf a_nxt = a_cur;
        if (more) a_nxt = ld_a_tile(A, ldA, r0, k0 + 32, lane);

        // read all four B tiles first, then an uninterrupted WMMA chain
        const bf16_t* bb = &Bsm[it & 1][0];
        v16bf b[4];
#pragma unroll
        for (int i = 0; i < 4; ++i) b[i] = ld_b_lds(bb, 16 * i, lane);
#pragma unroll
        for (int i = 0; i < 4; ++i) acc[i] = WMMA_BF16(a_cur, b[i], acc[i]);

        a_cur = a_nxt;
        __syncthreads();                       // everyone done before buffer reuse
    }

    const int n  = lane & 15;
    const int hf = lane >> 4;
    if (mode == 0) {
        bf16_t* out = (bf16_t*)outp;
#pragma unroll
        for (int i = 0; i < 4; ++i)
#pragma unroll
            for (int j = 0; j < 8; ++j)
                out[(size_t)(r0 + j + 8 * hf) * ldO + (c0 + 16 * i + n)] = (bf16_t)acc[i][j];
    } else if (mode == 1) {
        bf16_t* out = (bf16_t*)outp;
#pragma unroll
        for (int i = 0; i < 4; ++i)
#pragma unroll
            for (int j = 0; j < 8; ++j)
                out[(size_t)(c0 + 16 * i + n) * ldO + (r0 + j + 8 * hf)] = (bf16_t)acc[i][j];
    } else {
        float* out = (float*)outp;
#pragma unroll
        for (int i = 0; i < 4; ++i)
#pragma unroll
            for (int j = 0; j < 8; ++j)
                out[(size_t)(r0 + j + 8 * hf) * ldO + (c0 + 16 * i + n)] = acc[i][j];
    }
}

// ---------------------------------------------------------------------------
// RoPE, mirroring the reference's indexing (inv_freq[d % 32], pairs (2i,2i+1))
// ---------------------------------------------------------------------------
__global__ void rope_bf16(bf16_t* __restrict__ X, int ncols) {
    const int pairs_per_row = ncols >> 1;
    const int idx = blockIdx.x * blockDim.x + threadIdx.x;
    if (idx >= T_SEQ * pairs_per_row) return;
    const int t = idx / pairs_per_row;
    const int c = idx - t * pairs_per_row;
    const int d0 = 2 * c;
    const int dh0 = d0 & (HD - 1);
    const int dh1 = dh0 + 1;
    const float kLn = 0.28782313663f;     // ln(10000)/32
    float th0 = (float)t * __expf(-(float)(dh0 & 31) * kLn);
    float th1 = (float)t * __expf(-(float)(dh1 & 31) * kLn);
    size_t base = (size_t)t * ncols + d0;
    float x0 = (float)X[base];
    float x1 = (float)X[base + 1];
    X[base]     = (bf16_t)(x0 * __cosf(th0) - x1 * __sinf(th0));
    X[base + 1] = (bf16_t)(x1 * __cosf(th1) + x0 * __sinf(th1));
}

// ---------------------------------------------------------------------------
// Flash-style causal GQA attention. One wave per (head, 16-query block).
// Q: bf16 [T, 2048]; Kc: bf16 [T, 512]; Vt: bf16 [512, T]; O: bf16 [T, 2048].
// ---------------------------------------------------------------------------
__global__ __launch_bounds__(32)
void attn_fwd(const bf16_t* __restrict__ Q, const bf16_t* __restrict__ Kc,
              const bf16_t* __restrict__ Vt, bf16_t* __restrict__ O) {
    const int lane = threadIdx.x & 31;
    const int qb = blockIdx.x;
    const int h  = blockIdx.y;
    const int kvh = h >> 2;
    const int r0 = qb * 16;
    const int n  = lane & 15;
    const int hf = lane >> 4;

    __shared__ alignas(16) bf16_t Pl[16 * 32];

    v8f o[4] = {};
    float mrow[8], lrow[8];
#pragma unroll
    for (int j = 0; j < 8; ++j) { mrow[j] = -1e30f; lrow[j] = 0.f; }

    const float scale = 0.125f;       // 1/sqrt(64)
    const int nkb = (r0 + 16 + 31) >> 5;

    for (int kb = 0; kb < nkb; ++kb) {
        const int kbase = kb * 32;

        v8f s0 = {}, s1 = {};
#pragma unroll
        for (int dc = 0; dc < 2; ++dc) {
            const int k0 = dc * 32;
            v16bf a  = ld_a_tile(Q,  HID,   r0,         h * HD + k0,   lane);
            v16bf b0 = ld_b_tile(Kc, KVDIM, kbase,      kvh * HD + k0, lane);
            v16bf b1 = ld_b_tile(Kc, KVDIM, kbase + 16, kvh * HD + k0, lane);
            s0 = WMMA_BF16(a, b0, s0);
            s1 = WMMA_BF16(a, b1, s1);
        }

        float p0[8], p1[8];
#pragma unroll
        for (int j = 0; j < 8; ++j) {
            const int qrow = r0 + j + 8 * hf;
            const bool m0 = (kbase + n)      <= qrow;
            const bool m1 = (kbase + 16 + n) <= qrow;
            float v0 = m0 ? s0[j] * scale : -1e30f;
            float v1 = m1 ? s1[j] * scale : -1e30f;
            float bm = fmaxf(v0, v1);
#pragma unroll
            for (int off = 1; off < 16; off <<= 1)
                bm = fmaxf(bm, __shfl_xor(bm, off, 32));
            float newm = fmaxf(mrow[j], bm);
            float fac  = __expf(mrow[j] - newm);
            float e0 = m0 ? __expf(v0 - newm) : 0.f;
            float e1 = m1 ? __expf(v1 - newm) : 0.f;
            float ps = e0 + e1;
#pragma unroll
            for (int off = 1; off < 16; off <<= 1)
                ps += __shfl_xor(ps, off, 32);
            lrow[j] = lrow[j] * fac + ps;
            mrow[j] = newm;
#pragma unroll
            for (int i = 0; i < 4; ++i) o[i][j] *= fac;
            p0[j] = e0;
            p1[j] = e1;
        }

        // P (C layout) -> LDS -> A layout
#pragma unroll
        for (int j = 0; j < 8; ++j) {
            const int M = j + 8 * hf;
            Pl[M * 32 + n]      = (bf16_t)p0[j];
            Pl[M * 32 + 16 + n] = (bf16_t)p1[j];
        }
        __syncthreads();
        V16U pa;
        pa.h[0] = *reinterpret_cast<const v8bf*>(&Pl[n * 32 + hf * 16]);
        pa.h[1] = *reinterpret_cast<const v8bf*>(&Pl[n * 32 + hf * 16 + 8]);

#pragma unroll
        for (int i = 0; i < 4; ++i) {
            const bf16_t* vp = Vt + (size_t)(kvh * HD + i * 16 + n) * T_SEQ + kbase + hf * 16;
            v16bf vb = *reinterpret_cast<const v16bf*>(vp);
            o[i] = WMMA_BF16(pa.v, vb, o[i]);
        }
        __syncthreads();
    }

    float inv[8];
#pragma unroll
    for (int j = 0; j < 8; ++j) inv[j] = 1.f / lrow[j];
#pragma unroll
    for (int i = 0; i < 4; ++i)
#pragma unroll
        for (int j = 0; j < 8; ++j)
            O[(size_t)(r0 + j + 8 * hf) * HID + h * HD + i * 16 + n] = (bf16_t)(o[i][j] * inv[j]);
}

// ---------------------------------------------------------------------------
// Host-side launch
// ---------------------------------------------------------------------------
extern "C" void kernel_launch(void* const* d_in, const int* in_sizes, int n_in,
                              void* d_out, int out_size, void* d_ws, size_t ws_size,
                              hipStream_t stream) {
    (void)in_sizes; (void)n_in; (void)out_size; (void)ws_size;
    const float* x  = (const float*)d_in[0];
    const float* Wq = (const float*)d_in[1];
    const float* Wk = (const float*)d_in[2];
    const float* Wv = (const float*)d_in[3];
    const float* Wo = (const float*)d_in[4];
    float* out = (float*)d_out;

    char* ws = (char*)d_ws;
    const size_t MB = 1024 * 1024;
    bf16_t* xb   = (bf16_t*)(ws + 0 * MB);    // 8 MB  [2048,2048]
    bf16_t* wqb  = (bf16_t*)(ws + 8 * MB);    // 8 MB  [2048,2048]
    bf16_t* wkb  = (bf16_t*)(ws + 16 * MB);   // 2 MB  [512,2048]
    bf16_t* wvb  = (bf16_t*)(ws + 18 * MB);   // 2 MB  [512,2048]
    bf16_t* wob  = (bf16_t*)(ws + 20 * MB);   // 8 MB  [2048,2048]
    bf16_t* qbuf = (bf16_t*)(ws + 28 * MB);   // 8 MB  [2048,2048]
    bf16_t* kbuf = (bf16_t*)(ws + 36 * MB);   // 2 MB  [2048,512]
    bf16_t* vtb  = (bf16_t*)(ws + 38 * MB);   // 2 MB  [512,2048] (V transposed)
    bf16_t* ab   = (bf16_t*)(ws + 40 * MB);   // 8 MB  [2048,2048] attn out

    auto cast = [&](const float* src, bf16_t* dst, int nel) {
        cast_f32_bf16<<<(nel + 255) / 256, 256, 0, stream>>>(src, dst, nel);
    };
    cast(x,  xb,  HID * T_SEQ);
    cast(Wq, wqb, HID * HID);
    cast(Wk, wkb, KVDIM * HID);
    cast(Wv, wvb, KVDIM * HID);
    cast(Wo, wob, HID * HID);

    // Q = x*Wq^T;  K = x*Wk^T;  V^T = (x*Wv^T)^T
    wmma_gemm_bf16<<<dim3(T_SEQ / 64, HID / 64),   128, 0, stream>>>(xb, wqb, qbuf, HID, HID, HID, HID,   HID,   0);
    wmma_gemm_bf16<<<dim3(T_SEQ / 64, KVDIM / 64), 128, 0, stream>>>(xb, wkb, kbuf, HID, HID, HID, KVDIM, KVDIM, 0);
    wmma_gemm_bf16<<<dim3(T_SEQ / 64, KVDIM / 64), 128, 0, stream>>>(xb, wvb, vtb,  HID, HID, HID, T_SEQ, KVDIM, 1);

    rope_bf16<<<(T_SEQ * (HID / 2)   + 255) / 256, 256, 0, stream>>>(qbuf, HID);
    rope_bf16<<<(T_SEQ * (KVDIM / 2) + 255) / 256, 256, 0, stream>>>(kbuf, KVDIM);

    attn_fwd<<<dim3(T_SEQ / 16, NH), 32, 0, stream>>>(qbuf, kbuf, vtb, ab);

    // out = attn * Wo^T (f32 out)
    wmma_gemm_bf16<<<dim3(T_SEQ / 64, HID / 64), 128, 0, stream>>>(ab, wob, out, HID, HID, HID, HID, HID, 2);
}
